// Model_89197880803680
// MI455X (gfx1250) — matrix operand order repaired and checked by reading, not compiled, not run on hardware
//
#include <hip/hip_runtime.h>
#include <hip/hip_bf16.h>

// HTS hierarchical transformer, MI455X/gfx1250.
// Attention collapses exactly (broadcast tokens -> uniform softmax -> masked row
// select == v), so per block: h1=sp(x Wl^T); h2=sp(h1 Wl1^T); v=h2 Wv^T+bv;
// y=v Wo^T+bo; x+=y W20^T; finally out=x Wfc2^T.
// GEMMs: v_wmma_f32_16x16x32_bf16, double-buffered LDS fed by
// global_load_async_to_lds_b128 (ASYNCcnt pipeline).  K padded 168->192.

typedef __attribute__((ext_vector_type(16))) __bf16 v16bf;
typedef __attribute__((ext_vector_type(8)))  __bf16 v8bf;
typedef __attribute__((ext_vector_type(8)))  float  v8f;

#define HN   18900   // total series
#define HT   168     // time steps
#define HTP  192     // padded T (mult of 64)
#define HU   256     // d_model
#define HHOR 24      // horizon
#define HNB  3       // blocks

__device__ __forceinline__ float softplus_f(float z) {
    return (z > 20.0f) ? z : log1pf(__expf(z));
}

// low 32 bits of a generic shared-aperture address == group-segment LDS offset
__device__ __forceinline__ unsigned lds_off_of(const void* p) {
    return (unsigned)(uintptr_t)p;
}

__device__ __forceinline__ void wait_async0() {
#if __has_builtin(__builtin_amdgcn_s_wait_asynccnt)
    __builtin_amdgcn_s_wait_asynccnt(0);
#else
    asm volatile("s_wait_asynccnt 0" ::: "memory");
#endif
}

// Stage 64 contiguous bytes memory -> LDS without touching VGPR data paths.
// INST_OFFSET is added to both the global and the LDS address (ISA 08 §4.4).
__device__ __forceinline__ void stage64(const __bf16* g, unsigned lds) {
    asm volatile(
        "global_load_async_to_lds_b128 %0, %1, off\n\t"
        "global_load_async_to_lds_b128 %0, %1, off offset:16\n\t"
        "global_load_async_to_lds_b128 %0, %1, off offset:32\n\t"
        "global_load_async_to_lds_b128 %0, %1, off offset:48"
        :: "v"(lds), "v"(g) : "memory");
}

// ---------------------------------------------------------------------------
// Converters: fp32 -> bf16 with K padding
// ---------------------------------------------------------------------------
__global__ __launch_bounds__(256)
void cvt_pad_bf16(const float* __restrict__ src, __bf16* __restrict__ dst,
                  int rows, int K, int Kpad) {
    int idx = blockIdx.x * blockDim.x + threadIdx.x;
    if (idx >= rows * Kpad) return;
    int r = idx / Kpad, c = idx - r * Kpad;
    float v = (c < K) ? src[(size_t)r * K + c] : 0.0f;
    dst[idx] = (__bf16)v;
}

__global__ __launch_bounds__(256)
void init_x_kernel(const float* __restrict__ x, float* __restrict__ xf,
                   __bf16* __restrict__ xb) {
    int idx = blockIdx.x * blockDim.x + threadIdx.x;
    if (idx >= HN * HTP) return;
    int r = idx / HTP, c = idx - r * HTP;
    float v = (c < HT) ? x[(size_t)r * HT + c] : 0.0f;
    if (c < HT) xf[(size_t)r * HT + c] = v;
    xb[idx] = (__bf16)v;
}

// ---------------------------------------------------------------------------
// WMMA GEMM:  C[M x Nout] = epi( A[M x K] * B[Nout x K]^T ), bf16 in, f32 acc.
// EPI: 0 = softplus -> bf16, 1 = +bias -> bf16,
//      2 = residual: Cf += acc (fp32) and Cb = bf16(new), 3 = plain fp32.
// 128x128 block tile, 8 wave32 (2x4), wave tile 64x32 = 4x2 WMMA frags.
// Double-buffered LDS, async global->LDS copies overlapped with WMMA.
// Out-of-range rows/cols are CLAMPED at load (their C rows/cols are never
// stored), so the staging path is branch-free.
// ---------------------------------------------------------------------------
template <int EPI>
__global__ __launch_bounds__(256, 2)
void wmma_gemm(const __bf16* __restrict__ A, int ldA,
               const __bf16* __restrict__ B, int ldB,
               float* __restrict__ Cf, int ldCf,
               __bf16* __restrict__ Cb, int ldCb,
               const float* __restrict__ bias,
               int M, int Nout, int K) {
    constexpr int BM = 128, BN = 128, BK = 64, LDP = BK + 8;
    __shared__ __align__(16) __bf16 Asm[2][BM * LDP];
    __shared__ __align__(16) __bf16 Bsm[2][BN * LDP];

    const int tid  = threadIdx.x;
    const int lane = tid & 31;
    const int wave = tid >> 5;
    const int wM   = wave & 1;   // 2 waves along M
    const int wN   = wave >> 1;  // 4 waves along N
    const int rowBase = blockIdx.x * BM;
    const int colBase = blockIdx.y * BN;
    const int mlo  = lane & 15;
    const int hi16 = lane >> 4;

    v8f acc[4][2];
#pragma unroll
    for (int i = 0; i < 4; ++i)
#pragma unroll
        for (int j = 0; j < 2; ++j)
#pragma unroll
            for (int e = 0; e < 8; ++e) acc[i][j][e] = 0.0f;

    // staging map: thread -> (row, 32-element half-row); 64B per matrix each
    const int sRow = tid >> 1;
    const int sCol = (tid & 1) * 32;
    const __bf16* gA = A + (size_t)min(rowBase + sRow, M - 1)    * ldA + sCol;
    const __bf16* gB = B + (size_t)min(colBase + sRow, Nout - 1) * ldB + sCol;
    const unsigned lA[2] = { lds_off_of(&Asm[0][sRow * LDP + sCol]),
                             lds_off_of(&Asm[1][sRow * LDP + sCol]) };
    const unsigned lB[2] = { lds_off_of(&Bsm[0][sRow * LDP + sCol]),
                             lds_off_of(&Bsm[1][sRow * LDP + sCol]) };

    stage64(gA, lA[0]);
    stage64(gB, lB[0]);
    wait_async0();
    __syncthreads();

    int buf = 0;
    for (int k0 = 0; k0 < K; k0 += BK, buf ^= 1) {
        if (k0 + BK < K) {                    // prefetch next tile (other buffer)
            stage64(gA + k0 + BK, lA[buf ^ 1]);
            stage64(gB + k0 + BK, lB[buf ^ 1]);
        }
        const __bf16* Al = &Asm[buf][0];
        const __bf16* Bl = &Bsm[buf][0];
#pragma unroll
        for (int kk = 0; kk < BK; kk += 32) {
            // A frag (16x32, ISA 7.12.2): lanes 0-15: K 0-7|16-23, lanes 16-31:
            // K 8-15|24-31 of row M=lane&15.
            v16bf af[4];
#pragma unroll
            for (int i = 0; i < 4; ++i) {
                const __bf16* base = &Al[(wM * 64 + i * 16 + mlo) * LDP + kk];
                v8bf lo = *(const v8bf*)(base + hi16 * 8);
                v8bf hb = *(const v8bf*)(base + 16 + hi16 * 8);
                af[i] = __builtin_shufflevector(lo, hb, 0, 1, 2, 3, 4, 5, 6, 7,
                                                8, 9, 10, 11, 12, 13, 14, 15);
            }
            // B frag (32x16): lanes 0-15: K 0-7|16-23, lanes 16-31: K 8-15|24-31 of column lane&15
            v16bf bfr[2];
#pragma unroll
            for (int j = 0; j < 2; ++j) {
                const __bf16* base =
                    &Bl[(wN * 32 + j * 16 + mlo) * LDP + kk + hi16 * 8];
                v8bf lo = *(const v8bf*)(base);
                v8bf hb = *(const v8bf*)(base + 16);
                bfr[j] = __builtin_shufflevector(lo, hb, 0, 1, 2, 3, 4, 5, 6, 7,
                                                 8, 9, 10, 11, 12, 13, 14, 15);
            }
#pragma unroll
            for (int i = 0; i < 4; ++i)
#pragma unroll
                for (int j = 0; j < 2; ++j)
                    acc[i][j] = __builtin_amdgcn_wmma_f32_16x16x32_bf16(
                        false, af[i], false, bfr[j], (short)0, acc[i][j],
                        false, false);
        }
        wait_async0();     // next-tile async copies landed
        __syncthreads();   // all waves done reading + all copies visible
    }

    // Epilogue.  C/D layout: VGPR e holds M=e (lanes 0-15) / M=e+8 (16-31),
    // N = lane & 15.
    const int rOff = hi16 * 8;
#pragma unroll
    for (int i = 0; i < 4; ++i) {
#pragma unroll
        for (int j = 0; j < 2; ++j) {
            int col = colBase + wN * 32 + j * 16 + mlo;
#pragma unroll
            for (int e = 0; e < 8; ++e) {
                int row = rowBase + wM * 64 + i * 16 + rOff + e;
                if (row < M && col < Nout) {
                    float z = acc[i][j][e];
                    if (EPI == 0) {
                        Cb[(size_t)row * ldCb + col] = (__bf16)softplus_f(z);
                    } else if (EPI == 1) {
                        Cb[(size_t)row * ldCb + col] = (__bf16)(z + bias[col]);
                    } else if (EPI == 2) {
                        float s = Cf[(size_t)row * ldCf + col] + z;
                        Cf[(size_t)row * ldCf + col] = s;
                        Cb[(size_t)row * ldCb + col] = (__bf16)s;
                    } else {
                        Cf[(size_t)row * ldCf + col] = z;
                    }
                }
            }
        }
    }
}

// ---------------------------------------------------------------------------
extern "C" void kernel_launch(void* const* d_in, const int* in_sizes, int n_in,
                              void* d_out, int out_size, void* d_ws, size_t ws_size,
                              hipStream_t stream) {
    const float* x    = (const float*)d_in[0];
    const float* Wl   = (const float*)d_in[1];   // (3,256,168)
    const float* Wl1  = (const float*)d_in[2];   // (3,256,256)
    // d_in[3..6] = Wq,bq,Wk,bk : eliminated exactly by the math
    const float* Wv   = (const float*)d_in[7];   // (3,256,256)
    const float* bv   = (const float*)d_in[8];   // (3,256)
    const float* Wo   = (const float*)d_in[9];   // (3,256,256)
    const float* bo   = (const float*)d_in[10];  // (3,256)
    const float* W20  = (const float*)d_in[11];  // (3,168,256)
    const float* Wfc2 = (const float*)d_in[12];  // (24,168)
    float* out = (float*)d_out;

    // workspace carve-up (256B aligned)
    size_t off = 0;
    auto take = [&](size_t bytes) {
        size_t p = off;
        off = (off + bytes + 255) & ~(size_t)255;
        return p;
    };
    char* ws = (char*)d_ws;
    float*  xf   = (float*)(ws + take((size_t)HN * HT * 4));
    __bf16* xb   = (__bf16*)(ws + take((size_t)HN * HTP * 2));
    __bf16* bufA = (__bf16*)(ws + take((size_t)HN * HU * 2));
    __bf16* bufB = (__bf16*)(ws + take((size_t)HN * HU * 2));
    __bf16* wlB   = (__bf16*)(ws + take((size_t)HNB * HU * HTP * 2));
    __bf16* wl1B  = (__bf16*)(ws + take((size_t)HNB * HU * HU * 2));
    __bf16* wvB   = (__bf16*)(ws + take((size_t)HNB * HU * HU * 2));
    __bf16* woB   = (__bf16*)(ws + take((size_t)HNB * HU * HU * 2));
    __bf16* w20B  = (__bf16*)(ws + take((size_t)HNB * HT * HU * 2));
    __bf16* wfc2B = (__bf16*)(ws + take((size_t)HHOR * HTP * 2));

    auto cvt = [&](const float* s, __bf16* d, int rows, int K, int Kp) {
        int total = rows * Kp;
        cvt_pad_bf16<<<(total + 255) / 256, 256, 0, stream>>>(s, d, rows, K, Kp);
    };

    // weight + input conversion (every launch; inputs must not be cached)
    init_x_kernel<<<((HN * HTP) + 255) / 256, 256, 0, stream>>>(x, xf, xb);
    cvt(Wl,   wlB,   HNB * HU, HT, HTP);
    cvt(Wl1,  wl1B,  HNB * HU, HU, HU);
    cvt(Wv,   wvB,   HNB * HU, HU, HU);
    cvt(Wo,   woB,   HNB * HU, HU, HU);
    cvt(W20,  w20B,  HNB * HT, HU, HU);
    cvt(Wfc2, wfc2B, HHOR,     HT, HTP);

    const dim3 blk(256);
    auto grid = [](int M, int Nout) {
        return dim3((M + 127) / 128, (Nout + 127) / 128);
    };

    for (int b = 0; b < HNB; ++b) {
        const __bf16* wl   = wlB  + (size_t)b * HU * HTP;
        const __bf16* wl1  = wl1B + (size_t)b * HU * HU;
        const __bf16* wv   = wvB  + (size_t)b * HU * HU;
        const __bf16* wo   = woB  + (size_t)b * HU * HU;
        const __bf16* w20  = w20B + (size_t)b * HT * HU;
        const float*  bvp  = bv + b * HU;
        const float*  bop  = bo + b * HU;

        // h1 = softplus(x Wl^T)
        wmma_gemm<0><<<grid(HN, HU), blk, 0, stream>>>(
            xb, HTP, wl, HTP, nullptr, 0, bufA, HU, nullptr, HN, HU, HTP);
        // h2 = softplus(h1 Wl1^T)
        wmma_gemm<0><<<grid(HN, HU), blk, 0, stream>>>(
            bufA, HU, wl1, HU, nullptr, 0, bufB, HU, nullptr, HN, HU, HU);
        // v = h2 Wv^T + bv
        wmma_gemm<1><<<grid(HN, HU), blk, 0, stream>>>(
            bufB, HU, wv, HU, nullptr, 0, bufA, HU, bvp, HN, HU, HU);
        // y = v Wo^T + bo
        wmma_gemm<1><<<grid(HN, HU), blk, 0, stream>>>(
            bufA, HU, wo, HU, nullptr, 0, bufB, HU, bop, HN, HU, HU);
        // x += y W20^T   (updates fp32 master + bf16 padded copy)
        wmma_gemm<2><<<grid(HN, HT), blk, 0, stream>>>(
            bufB, HU, w20, HU, xf, HT, xb, HTP, nullptr, HN, HT, HU);
    }

    // out = x Wfc2^T  -> fp32 (N x 24)
    wmma_gemm<3><<<grid(HN, HHOR), blk, 0, stream>>>(
        xb, HTP, wfc2B, HTP, out, HHOR, nullptr, 0, nullptr, HN, HHOR, HTP);
}
